// MyModel_22273700397620
// MI455X (gfx1250) — compile-verified, matrix operand order checked by reading, not run on hardware
//
#include <hip/hip_runtime.h>

// ---------------------------------------------------------------------------
// MI455X (gfx1250) implementation of the prototype-clustering attention block.
//
// Shapes: B=16, C_IN=C_OUT=256, H=W=96 (N=9216), HEADS=8, HEAD_DIM=32,
//         BE=128, M=4 prototypes, D=256.
//
// Bandwidth-dominated problem: ~1 GB HBM traffic @ 23.3 TB/s vs 58 GFLOP of
// GEMM work on the bf16 WMMA path -> design for contiguous streams and keep
// the tensor cores busy without stalling on LDS.
//
//   1) convert x / Wf / Wv / Wp to bf16
//   2) WMMA GEMM (v_wmma_f32_16x16x32_bf16):
//        - A (weight) tile staged global->LDS via GLOBAL_LOAD_ASYNC_TO_LDS_B128
//          (ASYNCcnt, s_wait_asynccnt) -- rows are contiguous, no transpose
//        - X tile staged with a dword-load / ushort-scatter transpose
//        - LDS rows padded to 80 B -> each fragment = two aligned ds_load_b128
//        - all 8 B fragments preloaded, then 8 back-to-back WMMAs
//   3) fused cluster kernel: wave-per-pixel (lane = channel), shfl reductions,
//      sigmoid, argmax, ds_add_f32 LDS accumulation, global f32 atomics
//   4) finalize agg = (num + pooled/2304) / (den + 1)
//   5) dispatch mid[b][D][n] = bf16(sim[n] * agg[idx[n]][d])   (coalesced)
//   6) WMMA GEMM: Wp x mid + bp -> d_out [b][256][n] f32
// Assignment map (output #2) for be==0 written as int32 at d_out tail.
// ---------------------------------------------------------------------------

#define B_    16
#define CIN   256
#define COUT  256
#define HH    96
#define WW    96
#define NPIX  9216
#define HEADS 8
#define HD    32
#define BE_   128
#define MP    4
#define DTOT  256

typedef __attribute__((ext_vector_type(16))) __bf16    v16bf;
typedef __attribute__((ext_vector_type(8)))  float     v8f;
typedef __attribute__((ext_vector_type(4)))  unsigned  u32x4;

__device__ inline unsigned short f32_to_bf16(float f) {
  union { float f; unsigned u; } c; c.f = f;
  unsigned u = c.u;
  unsigned r = u + 0x7FFFu + ((u >> 16) & 1u);          // RNE
  if ((u & 0x7F800000u) == 0x7F800000u) r = u;          // inf/nan passthrough
  return (unsigned short)(r >> 16);
}

__device__ inline float wave_reduce_add(float v) {
  #pragma unroll
  for (int off = 16; off > 0; off >>= 1) v += __shfl_xor(v, off, 32);
  return v;
}

// ---------------------------------------------------------------------------
__global__ void cvt_f32_bf16(const float* __restrict__ src,
                             unsigned short* __restrict__ dst, long n) {
  long i = (long)blockIdx.x * blockDim.x + threadIdx.x;
  long stride = (long)gridDim.x * blockDim.x;
  for (; i < n; i += stride) dst[i] = f32_to_bf16(src[i]);
}

__global__ void zero_f32(float* __restrict__ p, long n) {
  long i = (long)blockIdx.x * blockDim.x + threadIdx.x;
  long stride = (long)gridDim.x * blockDim.x;
  for (; i < n; i += stride) p[i] = 0.0f;
}

// ---------------------------------------------------------------------------
// bf16 WMMA GEMM: Out[o][n] = sum_k W[o][k] * X[b][k][n] + bias[o]
// WG = 256 threads (8 waves). Tile 128(M) x 128(N), K staged 32 at a time.
// LDS pitch = 40 ushorts (80 B): 16-B aligned rows -> b128 fragment loads,
// and 20-dword stride -> conflict-free across the 16 rows of a fragment.
// mode 0: Out layout [be][n][32] f32 (be = b*8 + o/32)   (feat / value)
// mode 1: Out layout [b][o][n]  f32                      (final projection)
// ---------------------------------------------------------------------------
#define TM   128
#define TN   128
#define TK   32
#define LDR  40          // padded pitch in ushorts (80 bytes)

__global__ __launch_bounds__(256)
void gemm_bf16_wmma(const unsigned short* __restrict__ Wm,  // [O][K]
                    const unsigned short* __restrict__ Xm,  // [b][K][N]
                    const float* __restrict__ bias,         // [O]
                    float* __restrict__ Out,
                    int mode) {
  __shared__ __align__(16) unsigned short As[TM * LDR];
  __shared__ __align__(16) unsigned short Bs[TN * LDR];

  const int tid  = threadIdx.x;
  const int lane = tid & 31;
  const int wid  = tid >> 5;
  const int n0   = blockIdx.x * TN;
  const int o0   = blockIdx.y * TM;
  const int b    = blockIdx.z;
  const unsigned short* Xb = Xm + (size_t)b * CIN * NPIX;

  const unsigned ldsA = (unsigned)(uintptr_t)(&As[0]);
  const unsigned long long wbase = (unsigned long long)(uintptr_t)Wm;

  v8f acc[8] = {};

  for (int k0 = 0; k0 < CIN; k0 += TK) {
    // ---- stage A (TM x TK) asynchronously: global -> LDS, 16 B per lane ----
    // GLOBAL_LOAD_ASYNC_TO_LDS_B128, GVS mode: LDS-offset VGPR, 32-bit global
    // offset VGPR, 64-bit SGPR base.  Tracked by ASYNCcnt.
    #pragma unroll
    for (int i = 0; i < 2; ++i) {
      int e   = tid + i * 256;                 // 0..511 16-byte chunks
      int row = e >> 2;                        // 0..127
      int c4  = e & 3;                         // 16-B chunk within 64-B row
      unsigned goff = (unsigned)((((o0 + row) * CIN + k0) + c4 * 8) * 2);
      unsigned loff = ldsA + (unsigned)(row * (LDR * 2) + c4 * 16);
      asm volatile("global_load_async_to_lds_b128 %0, %1, %2"
                   :: "v"(loff), "v"(goff), "s"(wbase) : "memory");
    }

    // ---- stage B transposed: Bs[n_local][k] <- X[k0+k][n0+n_local] ----
    #pragma unroll
    for (int i = 0; i < 8; ++i) {
      int e  = tid + i * 256;                  // 0..2047 dwords
      int k  = e >> 6;                         // 0..31
      int n2 = e & 63;                         // dword column (2 pixels)
      unsigned v = *(const unsigned*)&Xb[(size_t)(k0 + k) * NPIX + n0 + n2 * 2];
      Bs[(n2 * 2    ) * LDR + k] = (unsigned short)(v & 0xFFFFu);
      Bs[(n2 * 2 + 1) * LDR + k] = (unsigned short)(v >> 16);
    }
    if (k0 + TK < CIN)  // warm next X K-block into cache
      __builtin_prefetch(&Xb[(size_t)(k0 + TK + (tid >> 3)) * NPIX + n0 + (tid & 7) * 16], 0, 0);

    asm volatile("s_wait_asynccnt 0x0" ::: "memory");   // A tile landed in LDS
    __syncthreads();                                    // B tile + cross-wave

    // ---- fragments: two aligned b128 LDS loads each (K pairs per ISA) ----
    const int kb = (lane < 16) ? 0 : 8;        // ushort offset of K-pair base
    union Frag { v16bf v; u32x4 q[2]; };

    Frag afr;
    {
      const unsigned short* basep = &As[(wid * 16 + (lane & 15)) * LDR];
      afr.q[0] = *(const u32x4*)(basep + kb);        // K = kb .. kb+7
      afr.q[1] = *(const u32x4*)(basep + kb + 16);   // K = kb+16 .. kb+23
    }
    Frag bfr[8];
    #pragma unroll
    for (int j = 0; j < 8; ++j) {
      const unsigned short* basep = &Bs[(j * 16 + (lane & 15)) * LDR];
      bfr[j].q[0] = *(const u32x4*)(basep + kb);
      bfr[j].q[1] = *(const u32x4*)(basep + kb + 16);
    }
    // ---- 8 back-to-back WMMAs (no per-op LDS waits) ----
    #pragma unroll
    for (int j = 0; j < 8; ++j)
      acc[j] = __builtin_amdgcn_wmma_f32_16x16x32_bf16(
          false, afr.v, false, bfr[j].v, (short)0, acc[j], false, false);

    __syncthreads();
  }

  // ---- epilogue: C/D layout: VGPR r -> row r (+8 for lanes 16..31) ----
  const int mbase = o0 + wid * 16;
  const int rext  = (lane >= 16) ? 8 : 0;
  #pragma unroll
  for (int j = 0; j < 8; ++j) {
    const int col = n0 + j * 16 + (lane & 15);
    #pragma unroll
    for (int r = 0; r < 8; ++r) {
      const int row = mbase + r + rext;
      const float v = acc[j][r] + bias[row];
      if (mode == 0) {
        const int be = b * HEADS + (row >> 5);
        Out[((size_t)be * NPIX + col) * HD + (row & 31)] = v;
      } else {
        Out[((size_t)b * COUT + row) * NPIX + col] = v;
      }
    }
  }
}

// ---------------------------------------------------------------------------
// Fused clustering pass.  Grid (BE, NSPLIT), block 256 (8 waves).
// Wave handles one pixel per iteration; lane = head-dim channel.
// ---------------------------------------------------------------------------
__global__ __launch_bounds__(256)
void cluster_assign(const float* __restrict__ feat,    // [BE][N][HD]
                    const float* __restrict__ value,   // [BE][N][HD]
                    const float* __restrict__ centers, // [MP][HD]
                    const float* __restrict__ alpha_p,
                    const float* __restrict__ beta_p,
                    float* __restrict__ simv,          // [BE][N]
                    int*   __restrict__ idxb,          // [BE][N]
                    float* __restrict__ aggnum,        // [BE][MP][HD]
                    float* __restrict__ aggden,        // [BE][MP]
                    float* __restrict__ pools,         // [BE][MP][HD]
                    int*   __restrict__ amap,          // [N]  (be==0)
                    int nsplit) {
  __shared__ float cn[MP * HD];
  __shared__ float l_num[MP * HD];
  __shared__ float l_den[MP];
  __shared__ float l_pool[MP * HD];

  const int tid  = threadIdx.x;
  const int lane = tid & 31;
  const int wid  = tid >> 5;
  const int be   = blockIdx.x;
  const int split = blockIdx.y;
  const float alpha = alpha_p[0];
  const float beta  = beta_p[0];

  if (tid < MP * HD) { l_num[tid] = 0.0f; l_pool[tid] = 0.0f; }
  if (tid < MP) l_den[tid] = 0.0f;
  if (tid < MP * HD) {                       // normalized centers (waves 0..3)
    float c  = centers[tid];
    float ss = wave_reduce_add(c * c);
    cn[tid] = c / fmaxf(sqrtf(ss), 1e-12f);
  }
  __syncthreads();

  const int nper = NPIX / nsplit;
  const int nbeg = split * nper;
  const int nend = nbeg + nper;

  for (int n = nbeg + wid; n < nend; n += 8) {
    const size_t pbase = ((size_t)be * NPIX + n) * HD + lane;
    float f  = feat[pbase];
    float ss = wave_reduce_add(f * f);
    float p  = f / fmaxf(sqrtf(ss), 1e-12f);

    float d0 = p * cn[0 * HD + lane];
    float d1 = p * cn[1 * HD + lane];
    float d2 = p * cn[2 * HD + lane];
    float d3 = p * cn[3 * HD + lane];
    #pragma unroll
    for (int off = 16; off > 0; off >>= 1) {
      d0 += __shfl_xor(d0, off, 32);
      d1 += __shfl_xor(d1, off, 32);
      d2 += __shfl_xor(d2, off, 32);
      d3 += __shfl_xor(d3, off, 32);
    }
    float s0 = 1.0f / (1.0f + __expf(-(beta + alpha * d0)));
    float s1 = 1.0f / (1.0f + __expf(-(beta + alpha * d1)));
    float s2 = 1.0f / (1.0f + __expf(-(beta + alpha * d2)));
    float s3 = 1.0f / (1.0f + __expf(-(beta + alpha * d3)));

    int bi = 0; float bs = s0;                // argmax, ties -> lowest index
    if (s1 > bs) { bs = s1; bi = 1; }
    if (s2 > bs) { bs = s2; bi = 2; }
    if (s3 > bs) { bs = s3; bi = 3; }

    float v = value[pbase];
    atomicAdd(&l_num[bi * HD + lane], bs * v);
    const int q = ((n / WW) / (HH / 2)) * 2 + ((n % WW) / (WW / 2));
    atomicAdd(&l_pool[q * HD + lane], v);
    if (lane == 0) {
      atomicAdd(&l_den[bi], bs);
      simv[(size_t)be * NPIX + n] = bs;
      idxb[(size_t)be * NPIX + n] = bi;
      if (be == 0) amap[n] = bi;
    }
  }
  __syncthreads();

  if (tid < MP * HD) {
    atomicAdd(&aggnum[be * MP * HD + tid], l_num[tid]);
    atomicAdd(&pools [be * MP * HD + tid], l_pool[tid]);
  }
  if (tid < MP) atomicAdd(&aggden[be * MP + tid], l_den[tid]);
}

// agg = (aggnum + pooled_mean) / (den + 1)
__global__ void agg_finalize(const float* __restrict__ aggnum,
                             const float* __restrict__ aggden,
                             const float* __restrict__ pools,
                             float* __restrict__ agg) {
  const int be = blockIdx.x;
  const int t  = threadIdx.x;            // 128 = MP*HD
  const int m  = t >> 5;
  const float vc  = pools[be * MP * HD + t] * (1.0f / 2304.0f);
  const float den = aggden[be * MP + m];
  agg[be * MP * HD + t] = (aggnum[be * MP * HD + t] + vc) / (den + 1.0f);
}

// mid[b][head*32+d][n] = bf16( simv[be][n] * agg[be][idx[be][n]][d] )
__global__ __launch_bounds__(256)
void dispatch_mid(const float* __restrict__ simv,
                  const int*   __restrict__ idxb,
                  const float* __restrict__ agg,
                  unsigned short* __restrict__ mid) {
  const int bd = blockIdx.x;             // 0..BE*HD-1
  const int be = bd >> 5;
  const int dl = bd & 31;
  const int b = be >> 3, head = be & 7;
  const float a0 = agg[be * MP * HD + 0 * HD + dl];
  const float a1 = agg[be * MP * HD + 1 * HD + dl];
  const float a2 = agg[be * MP * HD + 2 * HD + dl];
  const float a3 = agg[be * MP * HD + 3 * HD + dl];
  const size_t rowbase = ((size_t)b * DTOT + head * HD + dl) * NPIX;
  const size_t sbase   = (size_t)be * NPIX;
  for (int n = threadIdx.x; n < NPIX; n += 256) {
    const float s = simv[sbase + n];
    const int   m = idxb[sbase + n];
    const float a = (m == 0) ? a0 : (m == 1) ? a1 : (m == 2) ? a2 : a3;
    mid[rowbase + n] = f32_to_bf16(s * a);
  }
}

// ---------------------------------------------------------------------------
extern "C" void kernel_launch(void* const* d_in, const int* in_sizes, int n_in,
                              void* d_out, int out_size, void* d_ws, size_t ws_size,
                              hipStream_t stream) {
  const float* x       = (const float*)d_in[0];
  const float* Wf      = (const float*)d_in[1];
  const float* bf_     = (const float*)d_in[2];
  const float* Wv      = (const float*)d_in[3];
  const float* bv_     = (const float*)d_in[4];
  const float* Wp      = (const float*)d_in[5];
  const float* bp_     = (const float*)d_in[6];
  const float* alpha   = (const float*)d_in[7];
  const float* beta    = (const float*)d_in[8];
  const float* centers = (const float*)d_in[9];

  char* base = (char*)d_ws;
  size_t off = 0;
  auto take = [&](size_t bytes) -> char* {
    char* p = base + off;
    off += (bytes + 255) & ~(size_t)255;
    return p;
  };

  unsigned short* xbf   = (unsigned short*)take((size_t)B_ * CIN * NPIX * 2);
  unsigned short* wfbf  = (unsigned short*)take((size_t)COUT * CIN * 2);
  unsigned short* wvbf  = (unsigned short*)take((size_t)COUT * CIN * 2);
  unsigned short* wpbf  = (unsigned short*)take((size_t)COUT * DTOT * 2);
  float* featb  = (float*)take((size_t)BE_ * NPIX * HD * 4);
  float* valueb = (float*)take((size_t)BE_ * NPIX * HD * 4);
  float* simv   = (float*)take((size_t)BE_ * NPIX * 4);
  int*   idxb   = (int*)  take((size_t)BE_ * NPIX * 4);
  float* aggnum = (float*)take((size_t)BE_ * MP * HD * 4);   // contiguous w/
  float* aggden = (float*)take((size_t)BE_ * MP * 4);        // next two (256B
  float* pools  = (float*)take((size_t)BE_ * MP * HD * 4);   // multiples)
  float* agg    = (float*)take((size_t)BE_ * MP * HD * 4);
  unsigned short* mid = (unsigned short*)take((size_t)B_ * DTOT * NPIX * 2);

  float* out_f = (float*)d_out;
  int*   amap  = (int*)d_out + (size_t)B_ * COUT * NPIX;  // tail: [96x96] int32

  // 1) bf16 conversions
  cvt_f32_bf16<<<1024, 256, 0, stream>>>(x, xbf, (long)B_ * CIN * NPIX);
  cvt_f32_bf16<<<64, 256, 0, stream>>>(Wf, wfbf, (long)COUT * CIN);
  cvt_f32_bf16<<<64, 256, 0, stream>>>(Wv, wvbf, (long)COUT * CIN);
  cvt_f32_bf16<<<64, 256, 0, stream>>>(Wp, wpbf, (long)COUT * DTOT);

  // 2) feat / value GEMMs (mode 0 -> [be][n][32])
  dim3 gg(NPIX / TN, COUT / TM, B_);   // (72, 2, 16)
  gemm_bf16_wmma<<<gg, 256, 0, stream>>>(wfbf, xbf, bf_, featb, 0);
  gemm_bf16_wmma<<<gg, 256, 0, stream>>>(wvbf, xbf, bv_, valueb, 0);

  // 3) clustering (zero accumulators; aggnum/aggden/pools are contiguous)
  zero_f32<<<64, 256, 0, stream>>>(aggnum,
      (long)(BE_ * MP * HD + BE_ * MP + BE_ * MP * HD));
  dim3 gc(BE_, 8);
  cluster_assign<<<gc, 256, 0, stream>>>(featb, valueb, centers, alpha, beta,
                                         simv, idxb, aggnum, aggden, pools,
                                         amap, 8);
  agg_finalize<<<BE_, MP * HD, 0, stream>>>(aggnum, aggden, pools, agg);

  // 4) dispatch back to pixels in [b][D][n] bf16 layout
  dispatch_mid<<<BE_ * HD, 256, 0, stream>>>(simv, idxb, agg, mid);

  // 5) final projection straight into d_out (mode 1 -> [b][o][n])
  gemm_bf16_wmma<<<gg, 256, 0, stream>>>(wpbf, mid, bp_, out_f, 1);
}
